// Attention_34591666602794
// MI455X (gfx1250) — compile-verified, matrix operand order checked by reading, not run on hardware
//
#include <hip/hip_runtime.h>
#include <hip/hip_bf16.h>

// ---------------- constants ----------------
#define TT      4096
#define DIM     2048
#define NH      16
#define NKV     4
#define HD      128
#define KVD     (NKV*HD)     // 512
#define SCALE   0.08838834764831845f  // 1/sqrt(128)

typedef __attribute__((ext_vector_type(16))) __bf16 bf16x16;
typedef __attribute__((ext_vector_type(8)))  float  f32x8;

// ---- CDNA5 async global->LDS path (guarded; falls back to reg->ds staging) ----
#if defined(__has_builtin)
#if __has_builtin(__builtin_amdgcn_global_load_async_to_lds_b128)
#define HAS_ASYNC 1
#endif
#endif
#ifndef HAS_ASYNC
#define HAS_ASYNC 0
#endif

#if HAS_ASYNC
#if __has_builtin(__builtin_amdgcn_s_wait_asynccnt)
#define WAIT_ASYNC(n) __builtin_amdgcn_s_wait_asynccnt(n)
#else
#define WAIT_ASYNC(n) asm volatile("s_wait_asynccnt " #n ::: "memory")
#endif
#endif

// pointer types the async builtin expects (per hipcc diagnostic: int __vector(4))
typedef int v4i_vs __attribute__((vector_size(16)));
typedef __attribute__((address_space(1))) v4i_vs* as1_v4i;
typedef __attribute__((address_space(3))) v4i_vs* as3_v4i;

struct FragBits { uint4 lo, hi; };

// Build a 16x16x32 A/B fragment from two 16-byte chunks
// (v0..3 <- halves 0..7 @ p0, v4..7 <- halves 8..15 @ p1).
static __device__ __forceinline__ bf16x16 make_frag(const __hip_bfloat16* p0,
                                                    const __hip_bfloat16* p1) {
  FragBits u;
  u.lo = *reinterpret_cast<const uint4*>(p0);
  u.hi = *reinterpret_cast<const uint4*>(p1);
  return __builtin_bit_cast(bf16x16, u);
}

static __device__ __forceinline__ f32x8 wmma_bf16(bf16x16 a, bf16x16 b, f32x8 c) {
  return __builtin_amdgcn_wmma_f32_16x16x32_bf16(false, a, false, b, (short)0, c,
                                                 false, false);
}

// 16-byte global->LDS stage (async DMA if available, else via VGPRs)
static __device__ __forceinline__ void stage16(const __hip_bfloat16* g,
                                               __hip_bfloat16* l) {
#if HAS_ASYNC
  __builtin_amdgcn_global_load_async_to_lds_b128((as1_v4i)g, (as3_v4i)l, 0, 0);
#else
  *reinterpret_cast<uint4*>(l) = *reinterpret_cast<const uint4*>(g);
#endif
}

// ---------------- conversion kernels ----------------
__global__ void cvt_f32_bf16(const float* __restrict__ src,
                             __hip_bfloat16* __restrict__ dst, int n) {
  int i = blockIdx.x * blockDim.x + threadIdx.x;
  if (i < n) dst[i] = __float2bfloat16(src[i]);
}

// src[R][C] fp32 -> dst[C][R] bf16  (weight transpose: w[K][N] -> wT[N][K])
__global__ void cvt_f32_bf16_T(const float* __restrict__ src,
                               __hip_bfloat16* __restrict__ dst, int R, int C) {
  int i = blockIdx.x * blockDim.x + threadIdx.x;
  if (i >= R * C) return;
  int r = i / C, c = i - r * C;
  dst[(size_t)c * R + r] = __float2bfloat16(src[i]);
}

// RoPE + bf16 cast.  inp: [T][H][128] fp32, outb: same layout bf16.
__global__ void rope_cvt(const float* __restrict__ inp,
                         __hip_bfloat16* __restrict__ outb,
                         const float* __restrict__ cosb,
                         const float* __restrict__ sinb, int H) {
  int i = blockIdx.x * blockDim.x + threadIdx.x;
  int total = TT * H * (HD / 2);
  if (i >= total) return;
  int t = i / (H * 64);
  int rem = i - t * (H * 64);
  int hh = rem >> 6;
  int p  = rem & 63;
  float c = cosb[t * 64 + p];
  float s = sinb[t * 64 + p];
  size_t base = (size_t)t * (H * HD) + hh * HD + 2 * p;
  float x0 = inp[base], x1 = inp[base + 1];
  outb[base]     = __float2bfloat16(x0 * c - x1 * s);
  outb[base + 1] = __float2bfloat16(x0 * s + x1 * c);
}

// Vf [T][512] fp32 -> Vt [512][T] bf16 (per-kv-head dim-major)
__global__ void v_transpose(const float* __restrict__ Vf,
                            __hip_bfloat16* __restrict__ Vt) {
  int i = blockIdx.x * blockDim.x + threadIdx.x;
  if (i >= TT * KVD) return;
  int t = i >> 9;          // /512
  int c = i & (KVD - 1);   // kv-head*128 + d
  Vt[(size_t)c * TT + t] = __float2bfloat16(Vf[i]);
}

// ------- bf16 WMMA NT GEMM with async-LDS double buffering ------------------
// C[M][N] = A[M][K] * Bt[N][K]^T.  Block = 256 threads = 8 waves (4M x 2N),
// block tile 128x128, BK=32.  Tiles staged global->LDS (async DMA), fragments
// read via ds_load_b128 with 80B padded rows (bank-conflict-free).
#define BM 128
#define BN 128
#define LDST 40   // elements per 32-elem row (80 bytes)

__global__ __launch_bounds__(256) void gemm_bf16_nt(
    const __hip_bfloat16* __restrict__ A, const __hip_bfloat16* __restrict__ Bt,
    float* __restrict__ C, int M, int N, int K) {
  __shared__ __align__(16) __hip_bfloat16 sA[2][BM * LDST];
  __shared__ __align__(16) __hip_bfloat16 sB[2][BN * LDST];

  int tid  = threadIdx.x;
  int lane = tid & 31;
  int w    = tid >> 5;
  int half = lane >> 4;
  int n16  = lane & 15;
  int wm = w >> 1, wn = w & 1;          // 4 x 2 wave grid
  int mb = blockIdx.y * BM;
  int nb = blockIdx.x * BN;

  int r0  = tid >> 2;      // staging row (0..63), +64 on second rep
  int cc0 = tid & 3;       // 16B chunk within 64B row

  f32x8 acc[2][4];
  f32x8 z = 0.0f;
#pragma unroll
  for (int mi = 0; mi < 2; ++mi)
#pragma unroll
    for (int j = 0; j < 4; ++j) acc[mi][j] = z;

  int nk = K >> 5;

  auto stage = [&](int t, int buf) {
    int k0 = t << 5;
#pragma unroll
    for (int rep = 0; rep < 2; ++rep) {
      int r = r0 + rep * 64;
      stage16(A  + (size_t)(mb + r) * K + k0 + cc0 * 8, &sA[buf][r * LDST + cc0 * 8]);
      stage16(Bt + (size_t)(nb + r) * K + k0 + cc0 * 8, &sB[buf][r * LDST + cc0 * 8]);
    }
  };

  stage(0, 0);
  for (int t = 0; t < nk; ++t) {
    int cur = t & 1;
    if (t + 1 < nk) stage(t + 1, cur ^ 1);
#if HAS_ASYNC
    if (t + 1 < nk) { WAIT_ASYNC(4); } else { WAIT_ASYNC(0); }
#endif
    __syncthreads();

    bf16x16 af[2], bfr[4];
#pragma unroll
    for (int mi = 0; mi < 2; ++mi) {
      const __hip_bfloat16* p = &sA[cur][(wm * 32 + mi * 16 + n16) * LDST + half * 8];
      af[mi] = make_frag(p, p + 16);
    }
#pragma unroll
    for (int j = 0; j < 4; ++j) {
      const __hip_bfloat16* p = &sB[cur][(wn * 64 + j * 16 + n16) * LDST + half * 8];
      bfr[j] = make_frag(p, p + 16);
    }
#pragma unroll
    for (int mi = 0; mi < 2; ++mi)
#pragma unroll
      for (int j = 0; j < 4; ++j)
        acc[mi][j] = wmma_bf16(af[mi], bfr[j], acc[mi][j]);
    __syncthreads();
  }

#pragma unroll
  for (int mi = 0; mi < 2; ++mi)
#pragma unroll
    for (int j = 0; j < 4; ++j)
#pragma unroll
      for (int v = 0; v < 8; ++v)
        C[(size_t)(mb + wm * 32 + mi * 16 + v + 8 * half) * N +
          nb + wn * 64 + j * 16 + n16] = acc[mi][j][v];
}

// ---------------- flash attention ----------------
// 4 waves/block; wave w handles query tile qt = blockIdx.x*4+w for head blockIdx.y.
// Per 32-key block: 8 WMMA (Q·K^T) + online softmax + 8 WMMA (P·V).
__global__ __launch_bounds__(128) void flash_attn(
    const __hip_bfloat16* __restrict__ Qb,   // [T][16*128]
    const __hip_bfloat16* __restrict__ Kb,   // [T][4*128]
    const __hip_bfloat16* __restrict__ Vt,   // [4*128][T]
    const int* __restrict__ sid, float* __restrict__ Of /* [T][16*128] */) {
  __shared__ __align__(16) __hip_bfloat16 plds[4][16 * 32];

  int w = threadIdx.x >> 5;
  int lane = threadIdx.x & 31;
  int half = lane >> 4;
  int n16  = lane & 15;
  int qt = blockIdx.x * 4 + w;
  int h  = blockIdx.y;
  int hkv = h >> 2;
  int q0 = qt * 16;

  const __hip_bfloat16* qrow = Qb + (size_t)(q0 + n16) * (NH * HD) + h * HD;
  bf16x16 qf[4];
#pragma unroll
  for (int kk = 0; kk < 4; ++kk)
    qf[kk] = make_frag(qrow + kk * 32 + half * 8, qrow + kk * 32 + 16 + half * 8);

  int qsidv[8];
#pragma unroll
  for (int v = 0; v < 8; ++v) qsidv[v] = sid[q0 + v + 8 * half];
  int qsid = sid[q0];

  float m_i[8], l_i[8];
  f32x8 acc[8];
  f32x8 z = 0.0f;
#pragma unroll
  for (int v = 0; v < 8; ++v) { m_i[v] = -1e30f; l_i[v] = 0.0f; }
#pragma unroll
  for (int j = 0; j < 8; ++j) acc[j] = z;

  int kend = q0 + 16;
  int nkb = (kend + 31) >> 5;
  for (int kb = 0; kb < nkb; ++kb) {
    int k0 = kb * 32;
    int kprobe = k0 + 31 < kend - 1 ? k0 + 31 : kend - 1;
    if (sid[kprobe] < qsid) continue;   // block entirely before our sequence

    f32x8 s0 = z, s1 = z;
    const __hip_bfloat16* krow0 = Kb + (size_t)(k0 + n16) * KVD + hkv * HD;
    const __hip_bfloat16* krow1 = krow0 + (size_t)16 * KVD;
#pragma unroll
    for (int kk = 0; kk < 4; ++kk) {
      bf16x16 b0 = make_frag(krow0 + kk * 32 + half * 8, krow0 + kk * 32 + 16 + half * 8);
      s0 = wmma_bf16(qf[kk], b0, s0);
      bf16x16 b1 = make_frag(krow1 + kk * 32 + half * 8, krow1 + kk * 32 + 16 + half * 8);
      s1 = wmma_bf16(qf[kk], b1, s1);
    }

    int sid0 = sid[k0 + n16];
    int sid1 = sid[k0 + 16 + n16];

    float p0[8], p1[8], alpha[8];
#pragma unroll
    for (int v = 0; v < 8; ++v) {
      int qi = q0 + v + 8 * half;
      float v0 = ((k0 + n16) <= qi && sid0 == qsidv[v]) ? s0[v] * SCALE : -1e30f;
      float v1 = ((k0 + 16 + n16) <= qi && sid1 == qsidv[v]) ? s1[v] * SCALE : -1e30f;
      float mx = fmaxf(v0, v1);
#pragma unroll
      for (int off = 1; off < 16; off <<= 1) mx = fmaxf(mx, __shfl_xor(mx, off, 32));
      float mn = fmaxf(m_i[v], mx);
      alpha[v] = __expf(m_i[v] - mn);
      m_i[v] = mn;
      p0[v] = __expf(v0 - mn);
      p1[v] = __expf(v1 - mn);
      l_i[v] = l_i[v] * alpha[v] + p0[v] + p1[v];
    }
#pragma unroll
    for (int j = 0; j < 8; ++j)
#pragma unroll
      for (int v = 0; v < 8; ++v) acc[j][v] *= alpha[v];

    // stage P through LDS: D-layout -> A-layout, bf16
    __hip_bfloat16* pw = &plds[w][0];
#pragma unroll
    for (int v = 0; v < 8; ++v) {
      int m = v + 8 * half;
      pw[m * 32 + n16]      = __float2bfloat16(p0[v]);
      pw[m * 32 + 16 + n16] = __float2bfloat16(p1[v]);
    }
    __builtin_amdgcn_wave_barrier();   // keep ds store->load program order
    bf16x16 pf = make_frag(pw + n16 * 32 + half * 8, pw + n16 * 32 + 16 + half * 8);
    __builtin_amdgcn_wave_barrier();

#pragma unroll
    for (int j = 0; j < 8; ++j) {
      const __hip_bfloat16* vrow = Vt + (size_t)(hkv * HD + j * 16 + n16) * TT + k0;
      bf16x16 vf = make_frag(vrow + half * 8, vrow + 16 + half * 8);
      acc[j] = wmma_bf16(pf, vf, acc[j]);
    }
  }

  float linv[8];
#pragma unroll
  for (int v = 0; v < 8; ++v) {
    float ls = l_i[v];
#pragma unroll
    for (int off = 1; off < 16; off <<= 1) ls += __shfl_xor(ls, off, 32);
    linv[v] = 1.0f / ls;
  }
#pragma unroll
  for (int j = 0; j < 8; ++j)
#pragma unroll
    for (int v = 0; v < 8; ++v)
      Of[(size_t)(q0 + v + 8 * half) * (NH * HD) + h * HD + j * 16 + n16] =
          acc[j][v] * linv[v];
}

// ---------------- host side ----------------
extern "C" void kernel_launch(void* const* d_in, const int* in_sizes, int n_in,
                              void* d_out, int out_size, void* d_ws, size_t ws_size,
                              hipStream_t stream) {
  const float* x  = (const float*)d_in[0];
  const float* wq = (const float*)d_in[1];
  const float* wk = (const float*)d_in[2];
  const float* wv = (const float*)d_in[3];
  const float* wo = (const float*)d_in[4];
  const float* fc = (const float*)d_in[5];
  const float* fs = (const float*)d_in[6];
  const int*  sid = (const int*)d_in[7];
  float* out = (float*)d_out;

  char* ws = (char*)d_ws;
  size_t off = 0;
  __hip_bfloat16* xb  = (__hip_bfloat16*)(ws + off); off += (size_t)TT * DIM * 2;
  __hip_bfloat16* wqT = (__hip_bfloat16*)(ws + off); off += (size_t)DIM * DIM * 2;
  __hip_bfloat16* wkT = (__hip_bfloat16*)(ws + off); off += (size_t)KVD * DIM * 2;
  __hip_bfloat16* wvT = (__hip_bfloat16*)(ws + off); off += (size_t)KVD * DIM * 2;
  __hip_bfloat16* woT = (__hip_bfloat16*)(ws + off); off += (size_t)DIM * DIM * 2;
  float* Qf = (float*)(ws + off);                    off += (size_t)TT * DIM * 4;
  float* Kf = (float*)(ws + off);                    off += (size_t)TT * KVD * 4;
  float* Vf = (float*)(ws + off);                    off += (size_t)TT * KVD * 4;
  __hip_bfloat16* Qb = (__hip_bfloat16*)(ws + off);  off += (size_t)TT * DIM * 2;
  __hip_bfloat16* Kb = (__hip_bfloat16*)(ws + off);  off += (size_t)TT * KVD * 2;
  __hip_bfloat16* Vt = (__hip_bfloat16*)(ws + off);  off += (size_t)TT * KVD * 2;
  float* Of = Qf;                        // reuse: Qf dead after RoPE/cvt
  __hip_bfloat16* Ob = xb;               // reuse: xb dead after QKV GEMMs

  const int THR = 256;
  auto blks = [](size_t n, int t) { return (unsigned)((n + t - 1) / t); };

  // 1) bf16 conversions (weights transposed to [N][K])
  cvt_f32_bf16<<<blks((size_t)TT * DIM, THR), THR, 0, stream>>>(x, xb, TT * DIM);
  cvt_f32_bf16_T<<<blks((size_t)DIM * DIM, THR), THR, 0, stream>>>(wq, wqT, DIM, DIM);
  cvt_f32_bf16_T<<<blks((size_t)DIM * KVD, THR), THR, 0, stream>>>(wk, wkT, DIM, KVD);
  cvt_f32_bf16_T<<<blks((size_t)DIM * KVD, THR), THR, 0, stream>>>(wv, wvT, DIM, KVD);
  cvt_f32_bf16_T<<<blks((size_t)DIM * DIM, THR), THR, 0, stream>>>(wo, woT, DIM, DIM);

  // 2) QKV projections (WMMA bf16, fp32 accum, async-LDS staged)
  gemm_bf16_nt<<<dim3(DIM / BN, TT / BM), 256, 0, stream>>>(xb, wqT, Qf, TT, DIM, DIM);
  gemm_bf16_nt<<<dim3(KVD / BN, TT / BM), 256, 0, stream>>>(xb, wkT, Kf, TT, KVD, DIM);
  gemm_bf16_nt<<<dim3(KVD / BN, TT / BM), 256, 0, stream>>>(xb, wvT, Vf, TT, KVD, DIM);

  // 3) RoPE + bf16 cast; V transpose to [dim][T]
  rope_cvt<<<blks((size_t)TT * NH * 64, THR), THR, 0, stream>>>(Qf, Qb, fc, fs, NH);
  rope_cvt<<<blks((size_t)TT * NKV * 64, THR), THR, 0, stream>>>(Kf, Kb, fc, fs, NKV);
  v_transpose<<<blks((size_t)TT * KVD, THR), THR, 0, stream>>>(Vf, Vt);

  // 4) flash attention: 256 query tiles x 16 heads, 4 waves/block
  flash_attn<<<dim3(TT / 16 / 4, NH), 128, 0, stream>>>(Qb, Kb, Vt, sid, Of);

  // 5) output projection
  cvt_f32_bf16<<<blks((size_t)TT * DIM, THR), THR, 0, stream>>>(Of, Ob, TT * DIM);
  gemm_bf16_nt<<<dim3(DIM / BN, TT / BM), 256, 0, stream>>>(Ob, woT, out, TT, DIM, DIM);
}
